// LowBitMLP_43576738185537
// MI455X (gfx1250) — matrix operand order repaired and checked by reading, not compiled, not run on hardware
//
#include <hip/hip_runtime.h>
#include <hip/hip_bf16.h>

typedef __attribute__((ext_vector_type(8)))  _Float16 v8h;
typedef __attribute__((ext_vector_type(16))) _Float16 v16h;
typedef __attribute__((ext_vector_type(8)))  float    v8f;

#define LN_EPS 1e-5f
#define TERN_TH 0.1f

// ---------------------------------------------------------------------------
// Elementwise converts
// ---------------------------------------------------------------------------
__global__ __launch_bounds__(256) void cast_f32_to_f16_kernel(
    const float* __restrict__ in, _Float16* __restrict__ out, size_t n) {
  size_t i = (size_t)blockIdx.x * blockDim.x + threadIdx.x;
  if (i < n) out[i] = (_Float16)in[i];
}

__global__ __launch_bounds__(256) void ternarize_f32_to_f16_kernel(
    const float* __restrict__ in, _Float16* __restrict__ out, size_t n) {
  size_t i = (size_t)blockIdx.x * blockDim.x + threadIdx.x;
  if (i < n) {
    float w = in[i];
    float t = (fabsf(w) < TERN_TH) ? 0.0f : (w > 0.0f ? 1.0f : -1.0f);
    out[i] = (_Float16)t;
  }
}

// ---------------------------------------------------------------------------
// WMMA GEMM:  C[M x N] = f16( (A[M x K] * B[N x K]^T + bias[N]) * scale[N] )
// Block = 256 threads (8 waves). Block tile 128(M) x 256(N).
// Wave tile 64x64 = 4x4 grid of 16x16 WMMA tiles, K-step 32.
// 16 WMMAs per 16 global b128 loads per K-step.
// ---------------------------------------------------------------------------
__global__ __launch_bounds__(256) void gemm_tern_f16_kernel(
    const _Float16* __restrict__ A,   // M x K row-major (f16 activations)
    const _Float16* __restrict__ B,   // N x K row-major (f16 ternary weights)
    _Float16* __restrict__ C,         // M x N row-major (f16 pre-LN)
    const float* __restrict__ bias,   // [N]
    const float* __restrict__ scale,  // [N]
    int M, int N, int K) {
  const int lane  = threadIdx.x & 31;
  const int wid   = threadIdx.x >> 5;
  const int waveM = wid & 1;  // 0..1 -> 64-row strip
  const int waveN = wid >> 1; // 0..3 -> 64-col strip

  const int m0 = blockIdx.x * 128 + waveM * 64;
  const int n0 = blockIdx.y * 256 + waveN * 64;

  // A fragment addressing (16-bit A 16x32 layout, wave32):
  //   lane: row = lane&15, K-base = (lane&16)?8:0; v0-3 = kb..kb+7, v4-7 = kb+16..kb+23
  const int am  = lane & 15;
  const int akb = (lane & 16) ? 8 : 0;
  // B fragment addressing (16-bit B 32x16 layout, wave32):
  //   lane: col = lane&15, K-base = (lane&16)?8:0; v0-3 = kb2..kb2+7, v4-7 = kb2+16..kb2+23
  const int bn  = lane & 15;
  const int bkb = (lane & 16) ? 8 : 0;

  const _Float16* arow[4];
#pragma unroll
  for (int mi = 0; mi < 4; ++mi)
    arow[mi] = A + (size_t)(m0 + mi * 16 + am) * K + akb;
  const _Float16* brow[4];
#pragma unroll
  for (int ni = 0; ni < 4; ++ni)
    brow[ni] = B + (size_t)(n0 + ni * 16 + bn) * K + bkb;

  v8f acc[4][4];
#pragma unroll
  for (int mi = 0; mi < 4; ++mi)
#pragma unroll
    for (int ni = 0; ni < 4; ++ni) acc[mi][ni] = (v8f){};

  for (int k0 = 0; k0 < K; k0 += 32) {
    v16h af[4], bf[4];
#pragma unroll
    for (int mi = 0; mi < 4; ++mi) {
      v8h lo = *(const v8h*)(arow[mi] + k0);        // K = kb .. kb+7
      v8h hi = *(const v8h*)(arow[mi] + k0 + 16);   // K = kb+16 .. kb+23
      af[mi] = __builtin_shufflevector(lo, hi, 0, 1, 2, 3, 4, 5, 6, 7,
                                       8, 9, 10, 11, 12, 13, 14, 15);
    }
#pragma unroll
    for (int ni = 0; ni < 4; ++ni) {
      v8h lo = *(const v8h*)(brow[ni] + k0);        // K = kb2 .. kb2+7
      v8h hi = *(const v8h*)(brow[ni] + k0 + 16);   // K = kb2+16 .. kb2+23
      bf[ni] = __builtin_shufflevector(lo, hi, 0, 1, 2, 3, 4, 5, 6, 7,
                                       8, 9, 10, 11, 12, 13, 14, 15);
    }
#pragma unroll
    for (int mi = 0; mi < 4; ++mi)
#pragma unroll
      for (int ni = 0; ni < 4; ++ni)
        acc[mi][ni] = __builtin_amdgcn_wmma_f32_16x16x32_f16(
            /*neg_a=*/false, af[mi], /*neg_b=*/false, bf[ni],
            /*c_mod=*/(short)0, acc[mi][ni],
            /*reuse_a=*/false, /*reuse_b=*/false);
  }

  // Epilogue: C layout — VGPR i: lanes 0-15 -> row i, lanes 16-31 -> row i+8
  const int rbase = (lane & 16) ? 8 : 0;
  const int ccol  = lane & 15;
#pragma unroll
  for (int ni = 0; ni < 4; ++ni) {
    const int col = n0 + ni * 16 + ccol;
    const float bv = bias[col];
    const float sv = scale[col];
#pragma unroll
    for (int mi = 0; mi < 4; ++mi) {
#pragma unroll
      for (int i = 0; i < 8; ++i) {
        const int row = m0 + mi * 16 + rbase + i;
        C[(size_t)row * N + col] = (_Float16)((acc[mi][ni][i] + bv) * sv);
      }
    }
  }
}

// ---------------------------------------------------------------------------
// Row LayerNorm over f16 input (stats in f32). One block per row.
// ---------------------------------------------------------------------------
__global__ __launch_bounds__(256) void ln_relu_f16_kernel(
    const _Float16* __restrict__ in, _Float16* __restrict__ out,
    const float* __restrict__ g, const float* __restrict__ b, int D) {
  __shared__ float ssum[256];
  __shared__ float ssq[256];
  const int tid = threadIdx.x;
  const _Float16* p = in + (size_t)blockIdx.x * D;
  float sum = 0.f, sq = 0.f;
  for (int c = tid; c < D; c += 256) {
    float v = (float)p[c];
    sum += v;
    sq += v * v;
  }
  ssum[tid] = sum;
  ssq[tid] = sq;
  __syncthreads();
  for (int s = 128; s > 0; s >>= 1) {
    if (tid < s) {
      ssum[tid] += ssum[tid + s];
      ssq[tid] += ssq[tid + s];
    }
    __syncthreads();
  }
  const float mean = ssum[0] / (float)D;
  const float var  = ssq[0] / (float)D - mean * mean;
  const float rstd = rsqrtf(var + LN_EPS);
  _Float16* q = out + (size_t)blockIdx.x * D;
  for (int c = tid; c < D; c += 256) {
    float v = ((float)p[c] - mean) * rstd * g[c] + b[c];
    v = v > 0.f ? v : 0.f;  // relu
    q[c] = (_Float16)v;
  }
}

__global__ __launch_bounds__(256) void ln_f16_to_f32_kernel(
    const _Float16* __restrict__ in, float* __restrict__ out,
    const float* __restrict__ g, const float* __restrict__ b, int D) {
  __shared__ float ssum[256];
  __shared__ float ssq[256];
  const int tid = threadIdx.x;
  const _Float16* p = in + (size_t)blockIdx.x * D;
  float sum = 0.f, sq = 0.f;
  for (int c = tid; c < D; c += 256) {
    float v = (float)p[c];
    sum += v;
    sq += v * v;
  }
  ssum[tid] = sum;
  ssq[tid] = sq;
  __syncthreads();
  for (int s = 128; s > 0; s >>= 1) {
    if (tid < s) {
      ssum[tid] += ssum[tid + s];
      ssq[tid] += ssq[tid + s];
    }
    __syncthreads();
  }
  const float mean = ssum[0] / (float)D;
  const float var  = ssq[0] / (float)D - mean * mean;
  const float rstd = rsqrtf(var + LN_EPS);
  float* q = out + (size_t)blockIdx.x * D;
  for (int c = tid; c < D; c += 256) {
    q[c] = ((float)p[c] - mean) * rstd * g[c] + b[c];
  }
}

// ---------------------------------------------------------------------------
// Launch
// ---------------------------------------------------------------------------
extern "C" void kernel_launch(void* const* d_in, const int* in_sizes, int n_in,
                              void* d_out, int out_size, void* d_ws, size_t ws_size,
                              hipStream_t stream) {
  (void)in_sizes; (void)n_in; (void)out_size; (void)ws_size;

  const float* x    = (const float*)d_in[0];
  const float* w1   = (const float*)d_in[1];
  const float* b1   = (const float*)d_in[2];
  const float* s1   = (const float*)d_in[3];
  const float* w2   = (const float*)d_in[4];
  const float* b2   = (const float*)d_in[5];
  const float* s2   = (const float*)d_in[6];
  const float* ln1g = (const float*)d_in[7];
  const float* ln1b = (const float*)d_in[8];
  const float* outg = (const float*)d_in[9];
  const float* outb = (const float*)d_in[10];
  float* out = (float*)d_out;

  const size_t N = 8192, DIN = 2048, DH = 8192, DOUT = 2048;

  char* ws = (char*)d_ws;
  _Float16* xf   = (_Float16*)ws; ws += N * DIN * sizeof(_Float16);
  _Float16* w1f  = (_Float16*)ws; ws += DH * DIN * sizeof(_Float16);
  _Float16* w2f  = (_Float16*)ws; ws += DOUT * DH * sizeof(_Float16);
  _Float16* hpre = (_Float16*)ws; ws += N * DH * sizeof(_Float16);
  _Float16* hact = (_Float16*)ws; ws += N * DH * sizeof(_Float16);
  _Float16* opre = (_Float16*)ws;

  // 1) precision conversion / ternarization
  {
    size_t n = N * DIN;
    cast_f32_to_f16_kernel<<<(n + 255) / 256, 256, 0, stream>>>(x, xf, n);
  }
  {
    size_t n = DH * DIN;
    ternarize_f32_to_f16_kernel<<<(n + 255) / 256, 256, 0, stream>>>(w1, w1f, n);
  }
  {
    size_t n = DOUT * DH;
    ternarize_f32_to_f16_kernel<<<(n + 255) / 256, 256, 0, stream>>>(w2, w2f, n);
  }

  // 2) GEMM1: hpre = f16((xf * w1f^T + b1) * s1)    [8192 x 8192]
  gemm_tern_f16_kernel<<<dim3((int)(N / 128), (int)(DH / 256)), 256, 0, stream>>>(
      xf, w1f, hpre, b1, s1, (int)N, (int)DH, (int)DIN);

  // 3) hact = relu(layernorm(hpre))  -> f16
  ln_relu_f16_kernel<<<(int)N, 256, 0, stream>>>(hpre, hact, ln1g, ln1b, (int)DH);

  // 4) GEMM2: opre = f16((hact * w2f^T + b2) * s2)  [8192 x 2048]
  gemm_tern_f16_kernel<<<dim3((int)(N / 128), (int)(DOUT / 256)), 256, 0, stream>>>(
      hact, w2f, opre, b2, s2, (int)N, (int)DOUT, (int)DH);

  // 5) out = layernorm(opre)  -> f32
  ln_f16_to_f32_kernel<<<(int)N, 256, 0, stream>>>(opre, out, outg, outb, (int)DOUT);
}
